// LSTMForecaster_68143951118712
// MI455X (gfx1250) — matrix operand order repaired and checked by reading, not compiled, not run on hardware
//
#include <hip/hip_runtime.h>
#include <hip/hip_bf16.h>
#include <math.h>

typedef __attribute__((ext_vector_type(16))) __bf16 v16bf;
typedef __attribute__((ext_vector_type(8)))  __bf16 v8bf;
typedef __attribute__((ext_vector_type(8)))  float  v8f;

#define TSTEPS   336
#define DIN      8
#define NH1      96
#define NH2      64
#define HORIZON  24
#define A1S      136   // padded row stride (bf16) for A1 staging: [x(8) | h1(96) | pad]
#define A2S      168   // padded row stride (bf16) for A2 staging: [h1(96) | h2(64) | pad]
#define NTHREADS 320   // 10 waves: 0..5 = layer1 hidden blocks, 6..9 = layer2 hidden blocks

// Branch-free activations on the recurrent critical path:
// v_exp_f32 + v_rcp_f32, no exec-mask divergence, no lib calls.
__device__ __forceinline__ float sigm_(float x){
  return __builtin_amdgcn_rcpf(1.0f + __expf(-x));
}
__device__ __forceinline__ float tanh_(float x){
  float xc = fminf(fmaxf(x, -15.0f), 15.0f);   // avoids inf/inf, clamps to tanh saturation
  float e  = __expf(2.0f * xc);
  return (e - 1.0f) * __builtin_amdgcn_rcpf(e + 1.0f);
}

// Load one 16x32 bf16 WMMA A-operand chunk from an LDS staging buffer.
// A k order used here (contiguous, as packed B): lanes 0-15 hold row M=lane, K = c+0..15;
// lanes 16-31 hold row M=lane-16, K = c+16..31.
__device__ __forceinline__ v16bf ldA(const __bf16* buf, int stride, int k, int lane){
  const int row = lane & 15;
  const int off = (k << 5) + ((lane >> 4) << 4);
  const __bf16* p = buf + row * stride + off;
  v8bf lo = *(const v8bf*)(p);
  v8bf hi = *(const v8bf*)(p + 8);
  return __builtin_shufflevector(lo, hi, 0,1,2,3,4,5,6,7,8,9,10,11,12,13,14,15);
}

// ---------------- weight prep: pack into per-lane WMMA B-operand layout ----------------
// B k order here (contiguous, as ldA): lanes 0-15 hold N=lane, K = kbase+0..15; lanes 16-31 hold
// N=lane-16, K = kbase+16..31; element i of the lane's v16bf is K = kbase + (L>=16?16:0) + i.

// W1cat[384 x 128]: cols 0..7 = Wih1, cols 8..103 = Whh1, cols 104..127 = 0
// packed as [blk(6)][gate(4)][chunk(4)][lane(32)][16]
__global__ void pack_w1(const float* __restrict__ Wih1, const float* __restrict__ Whh1,
                        __bf16* __restrict__ outp){
  int e = blockIdx.x * blockDim.x + threadIdx.x;
  if (e >= 6*4*4*32*16) return;
  int i = e & 15;
  int L = (e >> 4) & 31;
  int k = (e >> 9) & 3;
  int g = (e >> 11) & 3;
  int w = e >> 13;
  int row = g*96 + w*16 + (L & 15);
  int col = (k << 5) + ((L >> 4) << 4) + i;
  float v = 0.0f;
  if (col < 8)        v = Wih1[row*8 + col];
  else if (col < 104) v = Whh1[row*96 + (col - 8)];
  outp[e] = (__bf16)v;
}

// W2cat[256 x 160]: cols 0..95 = Wih2, cols 96..159 = Whh2
// packed as [blk(4)][gate(4)][chunk(5)][lane(32)][16]
__global__ void pack_w2(const float* __restrict__ Wih2, const float* __restrict__ Whh2,
                        __bf16* __restrict__ outp){
  int e = blockIdx.x * blockDim.x + threadIdx.x;
  if (e >= 4*4*5*32*16) return;
  int i = e & 15;
  int L = (e >> 4) & 31;
  int r = e >> 9;
  int k = r % 5;
  int g = (r / 5) & 3;
  int w = r / 20;
  int row = g*64 + w*16 + (L & 15);
  int col = (k << 5) + ((L >> 4) << 4) + i;
  float v = (col < 96) ? Wih2[row*96 + col] : Whh2[row*64 + (col - 96)];
  outp[e] = (__bf16)v;
}

__global__ void fuse_bias(const float* __restrict__ bih1, const float* __restrict__ bhh1,
                          const float* __restrict__ bih2, const float* __restrict__ bhh2,
                          float* __restrict__ b1, float* __restrict__ b2){
  int t = blockIdx.x * blockDim.x + threadIdx.x;
  if (t < 4*NH1) b1[t] = bih1[t] + bhh1[t];
  if (t < 4*NH2) b2[t] = bih2[t] + bhh2[t];
}

// ---------------- main persistent kernel: one workgroup per 16-batch tile ----------------
__global__ __launch_bounds__(NTHREADS, 1)
void lstm_main(const float* __restrict__ x,
               const __bf16* __restrict__ w1p, const __bf16* __restrict__ w2p,
               const float* __restrict__ b1,  const float* __restrict__ b2,
               const float* __restrict__ fc1w, const float* __restrict__ fc1b,
               const float* __restrict__ fcow, const float* __restrict__ fcob,
               float* __restrict__ out){
  __shared__ __align__(16) __bf16 A1[16 * A1S];
  __shared__ __align__(16) __bf16 A2[16 * A2S];
  __shared__ float FH[16 * 64];

  const int tid  = threadIdx.x;
  const int lane = tid & 31;
  const int wv   = tid >> 5;
  const int b0   = blockIdx.x * 16;

  const bool isL1 = (wv < 6);
  const bool isL2 = (wv >= 6) && (wv < 10);
  const int  blk  = isL1 ? wv : (wv - 6);

  // Resident weights: L1 waves use wB[0..15] (4 gates x 4 K-chunks),
  // L2 waves use wB[0..19] (4 gates x 5 K-chunks).
  v16bf wB[20];
  float bi[4];
  if (isL1){
    #pragma unroll
    for (int g = 0; g < 4; ++g){
      #pragma unroll
      for (int k = 0; k < 4; ++k)
        wB[g*4 + k] = *(const v16bf*)(w1p + ((size_t)(((blk*4 + g)*4 + k)*32 + lane) * 16));
      bi[g] = b1[g*96 + blk*16 + (lane & 15)];
    }
  } else if (isL2){
    #pragma unroll
    for (int g = 0; g < 4; ++g){
      #pragma unroll
      for (int k = 0; k < 5; ++k)
        wB[g*5 + k] = *(const v16bf*)(w2p + ((size_t)(((blk*4 + g)*5 + k)*32 + lane) * 16));
      bi[g] = b2[g*64 + blk*16 + (lane & 15)];
    }
  }

  // Zero hidden-state staging (x columns 0..7 of A1 are written only by wave 0).
  for (int i = tid; i < 16 * A1S; i += NTHREADS){
    if ((i % A1S) >= 8) A1[i] = (__bf16)0.0f;
  }
  for (int i = tid; i < 16 * A2S; i += NTHREADS) A2[i] = (__bf16)0.0f;

  // Stage x[0]
  if (wv == 0){
    int row = lane >> 1, dp = (lane & 1) << 2;
    float4 x0 = *(const float4*)(x + (size_t)(b0 + row) * (TSTEPS * DIN) + dp);
    A1[row*A1S + dp + 0] = (__bf16)x0.x;
    A1[row*A1S + dp + 1] = (__bf16)x0.y;
    A1[row*A1S + dp + 2] = (__bf16)x0.z;
    A1[row*A1S + dp + 3] = (__bf16)x0.w;
  }
  __syncthreads();

  float c1s[8], c2s[8], hn[8];
  #pragma unroll
  for (int v = 0; v < 8; ++v){ c1s[v] = 0.0f; c2s[v] = 0.0f; hn[v] = 0.0f; }

  float4 xv = make_float4(0.f, 0.f, 0.f, 0.f);

  // t = 0..T: L1 runs for t<T producing h1[t]; L2 runs for t>=1 consuming h1[t-1].
  for (int t = 0; t <= TSTEPS; ++t){
    const bool doL1 = isL1 && (t < TSTEPS);
    const bool doL2 = isL2 && (t >= 1);

    // prefetch next x tile (overlaps the WMMA chains)
    if (wv == 0 && (t + 1) < TSTEPS){
      int row = lane >> 1, dp = (lane & 1) << 2;
      xv = *(const float4*)(x + (size_t)(b0 + row) * (TSTEPS * DIN) + (size_t)(t + 1) * DIN + dp);
    }

    if (doL1){
      v8f acc[4];
      #pragma unroll
      for (int g = 0; g < 4; ++g){
        #pragma unroll
        for (int i = 0; i < 8; ++i) acc[g][i] = bi[g];
      }
      #pragma unroll
      for (int k = 0; k < 4; ++k){
        v16bf a = ldA(A1, A1S, k, lane);
        #pragma unroll
        for (int g = 0; g < 4; ++g)
          acc[g] = __builtin_amdgcn_wmma_f32_16x16x32_bf16(false, a, false, wB[g*4 + k],
                                                           (short)0, acc[g], false, false);
      }
      #pragma unroll
      for (int v = 0; v < 8; ++v){
        float ig = sigm_(acc[0][v]);
        float fg = sigm_(acc[1][v]);
        float gg = tanh_(acc[2][v]);
        float og = sigm_(acc[3][v]);
        float c  = fg * c1s[v] + ig * gg;
        c1s[v] = c;
        hn[v]  = og * tanh_(c);
      }
    }
    if (doL2){
      v8f acc[4];
      #pragma unroll
      for (int g = 0; g < 4; ++g){
        #pragma unroll
        for (int i = 0; i < 8; ++i) acc[g][i] = bi[g];
      }
      #pragma unroll
      for (int k = 0; k < 5; ++k){
        v16bf a = ldA(A2, A2S, k, lane);
        #pragma unroll
        for (int g = 0; g < 4; ++g)
          acc[g] = __builtin_amdgcn_wmma_f32_16x16x32_bf16(false, a, false, wB[g*5 + k],
                                                           (short)0, acc[g], false, false);
      }
      #pragma unroll
      for (int v = 0; v < 8; ++v){
        float ig = sigm_(acc[0][v]);
        float fg = sigm_(acc[1][v]);
        float gg = tanh_(acc[2][v]);
        float og = sigm_(acc[3][v]);
        float c  = fg * c2s[v] + ig * gg;
        c2s[v] = c;
        hn[v]  = og * tanh_(c);
      }
    }

    __syncthreads();  // all A-operand reads complete before state is overwritten

    if (doL1){
      const int col = (lane & 15) + blk * 16;
      #pragma unroll
      for (int v = 0; v < 8; ++v){
        int row = v + ((lane >> 4) << 3);
        __bf16 hb = (__bf16)hn[v];
        A1[row*A1S + 8 + col] = hb;   // h1 for next L1 step
        A2[row*A2S + col]     = hb;   // h1 for next L2 step
      }
    }
    if (doL2){
      const int col = 96 + (lane & 15) + blk * 16;
      #pragma unroll
      for (int v = 0; v < 8; ++v){
        int row = v + ((lane >> 4) << 3);
        A2[row*A2S + col] = (__bf16)hn[v];
      }
    }
    if (wv == 0 && (t + 1) < TSTEPS){
      int row = lane >> 1, dp = (lane & 1) << 2;
      A1[row*A1S + dp + 0] = (__bf16)xv.x;
      A1[row*A1S + dp + 1] = (__bf16)xv.y;
      A1[row*A1S + dp + 2] = (__bf16)xv.z;
      A1[row*A1S + dp + 3] = (__bf16)xv.w;
    }

    __syncthreads();  // new state visible for next step
  }

  // ---- FC head: h2 final (16x64 bf16) sits in A2 cols 96..159 ----
  for (int idx = tid; idx < 16 * 64; idx += NTHREADS){
    int r = idx >> 6, j = idx & 63;
    float s = fc1b[j];
    #pragma unroll 8
    for (int k2 = 0; k2 < 64; ++k2)
      s += (float)A2[r*A2S + 96 + k2] * fc1w[j*64 + k2];
    FH[idx] = fmaxf(s, 0.0f);
  }
  __syncthreads();
  for (int idx = tid; idx < 16 * HORIZON; idx += NTHREADS){
    int r = idx / HORIZON, o = idx % HORIZON;
    float s = fcob[o];
    #pragma unroll 8
    for (int k2 = 0; k2 < 64; ++k2)
      s += FH[r*64 + k2] * fcow[o*64 + k2];
    out[(size_t)(b0 + r) * HORIZON + o] = s;
  }
}

extern "C" void kernel_launch(void* const* d_in, const int* in_sizes, int n_in,
                              void* d_out, int out_size, void* d_ws, size_t ws_size,
                              hipStream_t stream){
  const float* x    = (const float*)d_in[0];
  const float* Wih1 = (const float*)d_in[1];
  const float* Whh1 = (const float*)d_in[2];
  const float* bih1 = (const float*)d_in[3];
  const float* bhh1 = (const float*)d_in[4];
  const float* Wih2 = (const float*)d_in[5];
  const float* Whh2 = (const float*)d_in[6];
  const float* bih2 = (const float*)d_in[7];
  const float* bhh2 = (const float*)d_in[8];
  const float* fc1w = (const float*)d_in[9];
  const float* fc1b = (const float*)d_in[10];
  const float* fcow = (const float*)d_in[11];
  const float* fcob = (const float*)d_in[12];

  char* ws = (char*)d_ws;
  __bf16* w1p = (__bf16*)(ws);                         // 49152 bf16 = 98304 B
  __bf16* w2p = (__bf16*)(ws + 98304);                 // 40960 bf16 = 81920 B
  float*  b1  = (float*)(ws + 98304 + 81920);          // 384 f32
  float*  b2  = (float*)(ws + 98304 + 81920 + 1536);   // 256 f32

  hipLaunchKernelGGL(pack_w1, dim3(49152/256), dim3(256), 0, stream, Wih1, Whh1, w1p);
  hipLaunchKernelGGL(pack_w2, dim3(40960/256), dim3(256), 0, stream, Wih2, Whh2, w2p);
  hipLaunchKernelGGL(fuse_bias, dim3(2), dim3(256), 0, stream, bih1, bhh1, bih2, bhh2, b1, b2);
  hipLaunchKernelGGL(lstm_main, dim3(4096/16), dim3(NTHREADS), 0, stream,
                     x, w1p, w2p, b1, b2, fc1w, fc1b, fcow, fcob, (float*)d_out);
}
